// QRNN_48404281426012
// MI455X (gfx1250) — matrix operand order repaired and checked by reading, not compiled, not run on hardware
//
#include <hip/hip_runtime.h>
#include <stdint.h>
#include <math.h>

// QRNN: T=512, B=32, C=512, H=512, K=2, 2 layers, bidirectional.
// Gate convs as WMMA f16 GEMM (M=2048, N=16384, K=2*Cin) with double-buffered
// async global->LDS tiles, fused activation epilogue, bandwidth-bound scan.

typedef __attribute__((ext_vector_type(16))) _Float16 v16h;
typedef __attribute__((ext_vector_type(8)))  _Float16 v8h;
typedef __attribute__((ext_vector_type(8)))  float    v8f;

union AFragU { v16h v; v8h h[2]; _Float16 e[16]; };
union AccU   { v8f  v; float   e[8];  };

__device__ __forceinline__ void async_ld_b128(uint32_t lds_off, const void* g) {
  asm volatile("global_load_async_to_lds_b128 %0, %1, off"
               :: "v"(lds_off), "v"((uint64_t)(uintptr_t)g) : "memory");
}
__device__ __forceinline__ void wait_async0() {
  asm volatile("s_wait_asynccnt 0" ::: "memory");
}
__device__ __forceinline__ void wait_async4() {
  asm volatile("s_wait_asynccnt 4" ::: "memory");
}

// ---------------- weight convert: f32 (2,4,H,Cin,2) -> f16 A[d][2048][2*Cin]
// kk = c -> tap k=1 (x[t]); kk = Cin + c -> tap k=0 (x[t-1])
__global__ __launch_bounds__(256) void qrnn_cvt_w(const float* __restrict__ w,
                                                  _Float16* __restrict__ A,
                                                  int Cin, int n) {
  int i = blockIdx.x * 256 + threadIdx.x;
  if (i >= n) return;
  int c  = i % Cin;
  int r  = i / Cin;            // (d*4+g)*512 + h
  int h  = r & 511;
  int gg = (r >> 9) & 3;
  int d  = r >> 11;
  int KT = 2 * Cin;
  size_t dst = ((size_t)d * 2048 + (size_t)gg * 512 + h) * KT;
  A[dst + c]       = (_Float16)w[(size_t)i * 2 + 1];
  A[dst + Cin + c] = (_Float16)w[(size_t)i * 2 + 0];
}

// ---------------- layer-1 activations: x (T,B,C) f32 -> Bm[d][b][t][2C] f16
__global__ __launch_bounds__(256) void qrnn_prep_x1(const float* __restrict__ x,
                                                    _Float16* __restrict__ Bm) {
  int i = blockIdx.x * 256 + threadIdx.x;   // 2^24 total (d,b,t,c)
  int c = i & 511;
  int t = (i >> 9) & 511;
  int b = (i >> 18) & 31;
  int d = (i >> 23) & 1;
  float v1, v0;
  if (d == 0) {
    v1 = x[((size_t)t * 32 + b) * 512 + c];
    v0 = t ? x[((size_t)(t - 1) * 32 + b) * 512 + c] : 0.0f;
  } else {  // time-reversed input
    v1 = x[((size_t)(511 - t) * 32 + b) * 512 + c];
    v0 = t ? x[((size_t)(512 - t) * 32 + b) * 512 + c] : 0.0f;
  }
  size_t row = (size_t)d * 33554432 + ((size_t)b * 512 + t) * 1024;
  Bm[row + c]       = (_Float16)v1;
  Bm[row + 512 + c] = (_Float16)v0;
}

// ---------------- layer-2 activations from staged layer-1 output Hs[b][t][1024]
__global__ __launch_bounds__(256) void qrnn_prep_x2(const _Float16* __restrict__ Hs,
                                                    _Float16* __restrict__ Bm) {
  int i = blockIdx.x * 256 + threadIdx.x;   // 2^25 total (d,b,t,c)
  int c = i & 1023;
  int t = (i >> 10) & 511;
  int b = (i >> 19) & 31;
  int d = (i >> 24) & 1;
  size_t hb = (size_t)b * 512;
  _Float16 v1, v0;
  if (d == 0) {
    v1 = Hs[(hb + t) * 1024 + c];
    v0 = t ? Hs[(hb + t - 1) * 1024 + c] : (_Float16)0.0f;
  } else {
    v1 = Hs[(hb + (511 - t)) * 1024 + c];
    v0 = t ? Hs[(hb + (512 - t)) * 1024 + c] : (_Float16)0.0f;
  }
  size_t row = (size_t)d * 33554432 + ((size_t)b * 512 + t) * 2048;
  Bm[row + c]        = v1;
  Bm[row + 1024 + c] = v0;
}

// ---------------- WMMA GEMM: P = A(2048 x KTOT) * B(KTOT x 16384) + bias,
// fused tanh/sigmoid, gates stored f16 as G[d][g][b][t][h]
__global__ __launch_bounds__(256) void qrnn_gemm_gates(
    const _Float16* __restrict__ A,    // [2][2048][KTOT]       (K-contiguous)
    const _Float16* __restrict__ Bm,   // [2][32][512][KTOT]    (K-contiguous)
    const float* __restrict__ bias,    // [2][4][512]
    _Float16* __restrict__ G,          // [2][4][32][512][512]
    int KTOT)
{
  __shared__ __align__(16) _Float16 As[2][128][32];   // 16 KB  [row][k]
  __shared__ __align__(16) _Float16 Bs[2][128][40];   // 20 KB  [col][k] + pad

  const int tid  = threadIdx.x;
  const int lane = tid & 31;
  const int wv   = tid >> 5;     // 8 waves
  const int wm   = wv & 3;       // 4 waves across M (32 rows each)
  const int wn   = wv >> 2;      // 2 waves across N (64 cols each)
  const int lr   = lane & 15;
  const int lh   = lane >> 4;

  const int row0 = blockIdx.x * 128;
  const int col0 = blockIdx.y * 128;           // N index = b*512 + t
  const int d    = blockIdx.z;
  const int bb   = col0 >> 9;
  const int t0   = col0 & 511;

  const _Float16* Ad = A  + (size_t)d * 2048 * KTOT;
  const _Float16* Bd = Bm + (size_t)d * 33554432 + ((size_t)bb * 512 + t0) * KTOT;

  v8f acc[2][4];
  #pragma unroll
  for (int mi = 0; mi < 2; ++mi)
    #pragma unroll
    for (int ni = 0; ni < 4; ++ni)
      acc[mi][ni] = (v8f){0.f,0.f,0.f,0.f,0.f,0.f,0.f,0.f};

  auto issue_tile = [&](int it, int buf) {
    int kk0 = it * 32;
    #pragma unroll
    for (int q = 0; q < 2; ++q) {              // A tile 128x32: 512 x 16B
      int ch = tid + q * 256;
      int r = ch >> 2, k8 = (ch & 3) * 8;
      async_ld_b128((uint32_t)(uintptr_t)&As[buf][r][k8],
                    Ad + (size_t)(row0 + r) * KTOT + kk0 + k8);
    }
    #pragma unroll
    for (int q = 0; q < 2; ++q) {              // B tile 128 cols x 32 k
      int ch = tid + q * 256;
      int cl = ch >> 2, k8 = (ch & 3) * 8;
      async_ld_b128((uint32_t)(uintptr_t)&Bs[buf][cl][k8],
                    Bd + (size_t)cl * KTOT + kk0 + k8);
    }
  };

  const int ntiles = KTOT >> 5;
  issue_tile(0, 0);
  for (int it = 0; it < ntiles; ++it) {
    if (it + 1 < ntiles) { issue_tile(it + 1, (it + 1) & 1); wait_async4(); }
    else                 { wait_async0(); }
    __syncthreads();

    const int buf = it & 1;
    AFragU af[2];
    #pragma unroll
    for (int mi = 0; mi < 2; ++mi) {           // A frag: K {0-7,16-23}/{8-15,24-31}
      int row = wm * 32 + mi * 16 + lr;
      af[mi].h[0] = *(const v8h*)&As[buf][row][lh * 8];
      af[mi].h[1] = *(const v8h*)&As[buf][row][16 + lh * 8];
    }
    AFragU bf[4];
    #pragma unroll
    for (int ni = 0; ni < 4; ++ni) {           // B frag: K {0-7,16-23}/{8-15,24-31}
      int col = wn * 64 + ni * 16 + lr;
      bf[ni].h[0] = *(const v8h*)&Bs[buf][col][lh * 8];
      bf[ni].h[1] = *(const v8h*)&Bs[buf][col][16 + lh * 8];
    }
    #pragma unroll
    for (int mi = 0; mi < 2; ++mi)
      #pragma unroll
      for (int ni = 0; ni < 4; ++ni)
        acc[mi][ni] = __builtin_amdgcn_wmma_f32_16x16x32_f16(
            false, af[mi].v, false, bf[ni].v, (short)0, acc[mi][ni], false, false);
    __syncthreads();
  }

  // Epilogue: bias + tanh (g==0) / sigmoid, store f16 gates [g][b][t][h]
  #pragma unroll
  for (int mi = 0; mi < 2; ++mi) {
    #pragma unroll
    for (int ni = 0; ni < 4; ++ni) {
      AccU cc; cc.v = acc[mi][ni];
      int ncol = col0 + wn * 64 + ni * 16 + lr;
      int t = ncol & 511;
      #pragma unroll
      for (int v = 0; v < 8; ++v) {
        int m  = row0 + wm * 32 + mi * 16 + (lh << 3) + v;
        int gg = m >> 9, h = m & 511;
        float val = cc.e[v] + bias[(size_t)d * 2048 + m];
        val = (gg == 0) ? tanhf(val) : 1.0f / (1.0f + __expf(-val));
        G[(size_t)d * 33554432 +
          (((size_t)gg * 32 + bb) * 512 + t) * 512 + h] = (_Float16)val;
      }
    }
  }
}

// ---------------- recurrent scan: c = f*c + i*z; y = o*c. One thread/(b,h).
__global__ __launch_bounds__(256) void qrnn_scan(
    const _Float16* __restrict__ G,   // [2][4][32][512][512]
    _Float16* __restrict__ Hs,        // layer 0: staged output [b][t][1024]
    float* __restrict__ out,          // d_out
    int layer)
{
  int d = blockIdx.y;
  int n = blockIdx.x * 256 + threadIdx.x;   // 0..16383
  int b = n >> 9, h = n & 511;
  const _Float16* Gd = G + (size_t)d * 33554432;
  int c1 = d * 512 + h;
  float c = 0.0f;
  for (int t = 0; t < 512; ++t) {
    size_t o = ((size_t)b * 512 + t) * 512 + h;
    float z  = (float)Gd[o];
    float f  = (float)Gd[o + 8388608];
    float og = (float)Gd[o + 2 * 8388608];
    float ii = (float)Gd[o + 3 * 8388608];
    c = f * c + ii * z;
    float y = og * c;
    if (layer == 0) {
      Hs[((size_t)b * 512 + t) * 1024 + c1] = (_Float16)y;   // local time order
    } else {
      out[((size_t)t * 32 + b) * 1024 + c1] = y;             // (T,B,2H)
    }
    if ((d == 0 && t == 510) || (d == 1 && t == 0)) {        // h_n
      out[(size_t)16777216 + ((size_t)layer * 32 + b) * 1024 + c1] = y;
    }
  }
}

extern "C" void kernel_launch(void* const* d_in, const int* in_sizes, int n_in,
                              void* d_out, int out_size, void* d_ws, size_t ws_size,
                              hipStream_t stream) {
  const float* x  = (const float*)d_in[0];
  const float* w0 = (const float*)d_in[1];
  const float* b0 = (const float*)d_in[2];
  const float* w1 = (const float*)d_in[3];
  const float* b1 = (const float*)d_in[4];
  float* out = (float*)d_out;
  char* ws = (char*)d_ws;

  // workspace layout (bytes)
  _Float16* Bm = (_Float16*)(ws + 0);            // 134,217,728  (2 dirs, [b][t][2Cin])
  _Float16* Wc = (_Float16*)(ws + 134217728);    //  16,777,216  (f16 weights)
  _Float16* G  = (_Float16*)(ws + 150994944);    // 134,217,728  (f16 gates)
  _Float16* Hs = (_Float16*)(ws + 285212672);    //  33,554,432  (layer-1 output)

  dim3 blk(256);
  // ---- layer 1 (Cin=512, KTOT=1024)
  qrnn_cvt_w    <<<dim3(8192),        blk, 0, stream>>>(w0, Wc, 512, 2097152);
  qrnn_prep_x1  <<<dim3(65536),       blk, 0, stream>>>(x, Bm);
  qrnn_gemm_gates<<<dim3(16, 128, 2), blk, 0, stream>>>(Wc, Bm, b0, G, 1024);
  qrnn_scan     <<<dim3(64, 2),       blk, 0, stream>>>(G, Hs, out, 0);
  // ---- layer 2 (Cin=1024, KTOT=2048)
  qrnn_cvt_w    <<<dim3(16384),       blk, 0, stream>>>(w1, Wc, 1024, 4194304);
  qrnn_prep_x2  <<<dim3(131072),      blk, 0, stream>>>(Hs, Bm);
  qrnn_gemm_gates<<<dim3(16, 128, 2), blk, 0, stream>>>(Wc, Bm, b1, G, 2048);
  qrnn_scan     <<<dim3(64, 2),       blk, 0, stream>>>(G, Hs, out, 1);
}
